// Pos_emb_10531259810384
// MI455X (gfx1250) — matrix operand order repaired and checked by reading, not compiled, not run on hardware
//
#include <hip/hip_runtime.h>

// ---------------- problem constants ----------------
#define Bz   32
#define Sq   513          // S = M+1
#define Tt   (Bz * Sq)    // 16416 tokens, divisible by 16
#define Ee   512
#define Hh   8
#define DhD  64
#define Mm   512
#define Ff   768
#define DFFd 2048
#define SPp  544          // padded seq length (multiple of 32)
#define BH   (Bz * Hh)    // 256

typedef __attribute__((ext_vector_type(16))) __bf16 v16bf;
typedef __attribute__((ext_vector_type(8)))  float  v8f;
typedef __attribute__((ext_vector_type(4)))  int    v4i_;

union BF16Frag { v16bf v; unsigned int u[8]; uint4 q[2]; unsigned short s[16]; };

#if defined(__HIP_DEVICE_COMPILE__) && __has_builtin(__builtin_amdgcn_global_load_async_to_lds_b128)
#define HAVE_ASYNC_LDS 1
#else
#define HAVE_ASYNC_LDS 0
#endif

#if HAVE_ASYNC_LDS
typedef __attribute__((address_space(1))) v4i_* gptr_v4;
typedef __attribute__((address_space(3))) v4i_* lptr_v4;
static __device__ __forceinline__ void async_wait0() {
#if __has_builtin(__builtin_amdgcn_s_wait_asynccnt)
  __builtin_amdgcn_s_wait_asynccnt(0);
#else
  asm volatile("s_wait_asynccnt 0x0" ::: "memory");
#endif
}
#endif

static __device__ __forceinline__ unsigned short f2bf(float f) {
  unsigned int u = __float_as_uint(f);
  u += 0x7FFFu + ((u >> 16) & 1u);   // round-to-nearest-even
  return (unsigned short)(u >> 16);
}

static __device__ __forceinline__ v8f wmma_bf16(const BF16Frag& a, const BF16Frag& b, v8f c) {
  return __builtin_amdgcn_wmma_f32_16x16x32_bf16(false, a.v, false, b.v, (short)0, c, false, false);
}

// ---------------- fp32 -> bf16 weight conversion ----------------
__global__ __launch_bounds__(256) void f2bf_kernel(const float* __restrict__ in,
                                                   unsigned short* __restrict__ out, int n) {
  int i = blockIdx.x * 256 + threadIdx.x;
  if (i < n) out[i] = f2bf(in[i]);
}

// ---------------- embedding + span fill ----------------
__global__ __launch_bounds__(128) void embed_kernel(
    const int* __restrict__ pos1, const int* __restrict__ pos2,
    const float* __restrict__ final_base, const float* __restrict__ emb,
    float* __restrict__ xf, unsigned short* __restrict__ xh)
{
  const int row = blockIdx.x;          // 0..T-1
  const int b = row / Sq, s = row - b * Sq;
  __shared__ int sid;
  if (threadIdx.x == 0) {
    int best = -1;
    if (s > 0) {
      const int k = s - 1;
      const int* p = pos1 + (size_t)b * Mm * 2;
      for (int j = 0; j < Mm; ++j) {
        const int st = p[2 * j], en = p[2 * j + 1];
        if (en == 0) break;              // prefix of valid spans
        if (st <= k && k < en) best = j; // later spans overwrite -> max index
      }
    }
    sid = best;
  }
  __syncthreads();
  const int span = sid;
  for (int e = threadIdx.x; e < Ee; e += 128) {
    float v;
    if (s == 0) {
      v = emb[Ee + e];                   // cls = emb_table[1]
    } else if (span >= 0) {
      const int idx = pos2[(size_t)b * Mm + span];
      const float ev = (idx == 0) ? 0.0f : emb[(size_t)idx * Ee + e]; // padding_idx=0
      const float freq = expf(-((float)((e >> 1) << 1) / (float)Ee) * 9.210340371976184f);
      const float ang = (float)span * freq;
      v = ev + ((e & 1) ? cosf(ang) : sinf(ang));
    } else {
      v = final_base[((size_t)b * Mm + (s - 1)) * Ee + e];
    }
    xf[(size_t)row * Ee + e] = v;
    xh[(size_t)row * Ee + e] = f2bf(v);
  }
}

// ---------------- GEMM epilogue helper ----------------
static __device__ __forceinline__ void gemm_epilogue(
    int t, int n, float val, int N, int mode, const float* res, float* outf,
    unsigned short* outh, unsigned short* qb, unsigned short* kb_, unsigned short* vtb)
{
  if (mode == 0) {
    outf[(size_t)t * N + n] = val + res[(size_t)t * N + n];
  } else if (mode == 1) {
    outh[(size_t)t * N + n] = f2bf(fmaxf(val, 0.0f));
  } else if (mode == 2) {
    outf[(size_t)t * N + n] = tanhf(val);
  } else {
    const int bidx = t / Sq;
    const int l = t - bidx * Sq;
    if (n < Ee) {
      const int h = n >> 6, d = n & 63;
      qb[(((size_t)bidx * Hh + h) * Sq + l) * DhD + d] = f2bf(val * 0.125f);
    } else if (n < 2 * Ee) {
      const int nn = n - Ee, h = nn >> 6, d = nn & 63;
      kb_[(((size_t)bidx * Hh + h) * SPp + l) * DhD + d] = f2bf(val);
    } else {
      const int nn = n - 2 * Ee, h = nn >> 6, d = nn & 63;
      vtb[(((size_t)bidx * Hh + h) * DhD + d) * SPp + l] = f2bf(val);
    }
  }
}

// ---------------- generic WMMA GEMM: out = A[T,K] @ W[N,K]^T (+bias) ----------------
// Block: 256 threads = 8 waves; block tile = 16 rows x 256 cols (2 col tiles/wave).
// Whole 16 x Kdim A strip staged once into dynamic LDS (async-to-LDS when available).
__global__ __launch_bounds__(256) void gemm_kernel(
    const unsigned short* __restrict__ A, const unsigned short* __restrict__ W,
    const float* __restrict__ bias, const float* res,
    float* outf, unsigned short* __restrict__ outh,
    unsigned short* __restrict__ qb, unsigned short* __restrict__ kb_,
    unsigned short* __restrict__ vtb, int Kdim, int N, int mode)
{
  extern __shared__ unsigned int sA[];           // 16 rows x (Kdim/2) u32
  const int tid  = threadIdx.x;
  const int lane = tid & 31;
  const int wave = tid >> 5;
  const int m  = lane & 15;
  const int hi = lane >> 4;
  const int row0 = blockIdx.y << 4;
  const int col0 = (blockIdx.x << 8) + (wave << 5);   // 2 tiles of 16 per wave
  const int rowU32 = Kdim >> 1;

  // ---- stage full A strip: one barrier total ----
  {
    const int r = tid >> 4;
    const unsigned int* gsrc = (const unsigned int*)(A + (size_t)(row0 + r) * (size_t)Kdim);
    unsigned int* ldst = sA + r * rowU32;
    for (int j = (tid & 15) << 2; j < rowU32; j += 64) {
#if HAVE_ASYNC_LDS
      __builtin_amdgcn_global_load_async_to_lds_b128(
          (gptr_v4)(void*)(gsrc + j), (lptr_v4)(void*)(ldst + j), 0, 0);
#else
      *(uint4*)(ldst + j) = *(const uint4*)(gsrc + j);
#endif
    }
#if HAVE_ASYNC_LDS
    async_wait0();
#endif
  }
  __syncthreads();

  const unsigned int* wrow0 = (const unsigned int*)(W + (size_t)(col0 + m) * (size_t)Kdim);
  const unsigned int* wrow1 = (const unsigned int*)(W + (size_t)(col0 + 16 + m) * (size_t)Kdim);
  const unsigned int* sRow  = sA + m * rowU32;

  v8f acc0 = {}, acc1 = {};
  for (int kb2 = 0; kb2 < rowU32; kb2 += 16) {   // kb2 = k/2 (32 bf16 per step)
    BF16Frag af, bf0, bf1;
    af.q[0]  = *(const uint4*)(sRow + kb2 + (hi << 2));        // A 16x32 layout, b128 x2
    af.q[1]  = *(const uint4*)(sRow + kb2 + 8 + (hi << 2));
    bf0.q[0] = *(const uint4*)(wrow0 + kb2 + (hi << 2));       // B 32x16 layout, b128 x2
    bf0.q[1] = *(const uint4*)(wrow0 + kb2 + 8 + (hi << 2));
    bf1.q[0] = *(const uint4*)(wrow1 + kb2 + (hi << 2));
    bf1.q[1] = *(const uint4*)(wrow1 + kb2 + 8 + (hi << 2));
    acc0 = wmma_bf16(af, bf0, acc0);
    acc1 = wmma_bf16(af, bf1, acc1);
  }

  const int n0 = col0 + m;                       // D layout: col = lane&15
  const int n1 = col0 + 16 + m;
  const float bv0 = bias ? bias[n0] : 0.0f;
  const float bv1 = bias ? bias[n1] : 0.0f;
#pragma unroll
  for (int r = 0; r < 8; ++r) {                  // row = r + 8*(lane>>4)
    const int t = row0 + r + (hi << 3);
    gemm_epilogue(t, n0, acc0[r] + bv0, N, mode, res, outf, outh, qb, kb_, vtb);
    gemm_epilogue(t, n1, acc1[r] + bv1, N, mode, res, outf, outh, qb, kb_, vtb);
  }
}

// ---------------- flash attention: one wave per (bh, 16-row tile) ----------------
__global__ __launch_bounds__(32) void attn_kernel(
    const unsigned short* __restrict__ Q, const unsigned short* __restrict__ Kmat,
    const unsigned short* __restrict__ Vt, unsigned short* __restrict__ O)
{
  __shared__ unsigned short sP[16 * 32];
  const int lane = threadIdx.x;
  const int m = lane & 15, hi = lane >> 4;
  const int bh = blockIdx.y;
  const int b = bh >> 3, h = bh & 7;
  const int l0 = blockIdx.x << 4;
  int lrow = l0 + m; if (lrow > Sq - 1) lrow = Sq - 1;   // clamp OOB query rows

  const unsigned int* qrow = (const unsigned int*)(Q + ((size_t)bh * Sq + lrow) * DhD);
  BF16Frag qa0, qa1;                              // Q as A-matrix, d=0..31 / 32..63
  qa0.q[0] = *(const uint4*)(qrow + (hi << 2));
  qa0.q[1] = *(const uint4*)(qrow + 8 + (hi << 2));
  qa1.q[0] = *(const uint4*)(qrow + 16 + (hi << 2));
  qa1.q[1] = *(const uint4*)(qrow + 24 + (hi << 2));

  v8f o0 = {}, o1 = {}, o2 = {}, o3 = {};
  float mrun[8], lrun[8];
#pragma unroll
  for (int r = 0; r < 8; ++r) { mrun[r] = -1e30f; lrun[r] = 0.0f; }

  for (int c0 = 0; c0 < SPp; c0 += 32) {
    const unsigned int* kr0 = (const unsigned int*)(Kmat + ((size_t)bh * SPp + c0 + m) * DhD);
    const unsigned int* kr1 = (const unsigned int*)(Kmat + ((size_t)bh * SPp + c0 + 16 + m) * DhD);
    BF16Frag kf0, kf1;
    v8f s0 = {}, s1 = {};
    kf0.q[0] = *(const uint4*)(kr0 + (hi << 2));
    kf0.q[1] = *(const uint4*)(kr0 + 8 + (hi << 2));
    kf1.q[0] = *(const uint4*)(kr1 + (hi << 2));
    kf1.q[1] = *(const uint4*)(kr1 + 8 + (hi << 2));
    s0 = wmma_bf16(qa0, kf0, s0);
    s1 = wmma_bf16(qa0, kf1, s1);
    kf0.q[0] = *(const uint4*)(kr0 + 16 + (hi << 2));
    kf0.q[1] = *(const uint4*)(kr0 + 24 + (hi << 2));
    kf1.q[0] = *(const uint4*)(kr1 + 16 + (hi << 2));
    kf1.q[1] = *(const uint4*)(kr1 + 24 + (hi << 2));
    s0 = wmma_bf16(qa1, kf0, s0);
    s1 = wmma_bf16(qa1, kf1, s1);

    const bool ok0 = (c0 + m) < Sq;
    const bool ok1 = (c0 + 16 + m) < Sq;
#pragma unroll
    for (int r = 0; r < 8; ++r) {                 // online softmax per row
      float v0 = ok0 ? s0[r] : -1e30f;
      float v1 = ok1 ? s1[r] : -1e30f;
      float t = fmaxf(v0, v1);
      t = fmaxf(t, __shfl_xor(t, 1, 16));
      t = fmaxf(t, __shfl_xor(t, 2, 16));
      t = fmaxf(t, __shfl_xor(t, 4, 16));
      t = fmaxf(t, __shfl_xor(t, 8, 16));
      const float mnew = fmaxf(mrun[r], t);
      const float corr = __expf(mrun[r] - mnew);
      const float p0 = __expf(v0 - mnew);
      const float p1 = __expf(v1 - mnew);
      float rs = p0 + p1;
      rs += __shfl_xor(rs, 1, 16);
      rs += __shfl_xor(rs, 2, 16);
      rs += __shfl_xor(rs, 4, 16);
      rs += __shfl_xor(rs, 8, 16);
      lrun[r] = lrun[r] * corr + rs;
      mrun[r] = mnew;
      o0[r] *= corr; o1[r] *= corr; o2[r] *= corr; o3[r] *= corr;
      const int prow = r + (hi << 3);             // D-layout -> LDS row-major
      sP[(prow << 5) + m]      = f2bf(p0);
      sP[(prow << 5) + 16 + m] = f2bf(p1);
    }
    __syncthreads();                              // single wave: LDS in-order + fence
    BF16Frag pa;                                  // reload P in A-matrix layout
    const unsigned int* sprow = (const unsigned int*)(&sP[m << 5]);
    pa.q[0] = *(const uint4*)(sprow + (hi << 2));
    pa.q[1] = *(const uint4*)(sprow + 8 + (hi << 2));
    __syncthreads();

    {
      const unsigned int* vr0 = (const unsigned int*)(Vt + (((size_t)bh * DhD) + 0  + m) * SPp + c0);
      const unsigned int* vr1 = (const unsigned int*)(Vt + (((size_t)bh * DhD) + 16 + m) * SPp + c0);
      const unsigned int* vr2 = (const unsigned int*)(Vt + (((size_t)bh * DhD) + 32 + m) * SPp + c0);
      const unsigned int* vr3 = (const unsigned int*)(Vt + (((size_t)bh * DhD) + 48 + m) * SPp + c0);
      BF16Frag vf;
      vf.q[0] = *(const uint4*)(vr0 + (hi << 2));
      vf.q[1] = *(const uint4*)(vr0 + 8 + (hi << 2));
      o0 = wmma_bf16(pa, vf, o0);
      vf.q[0] = *(const uint4*)(vr1 + (hi << 2));
      vf.q[1] = *(const uint4*)(vr1 + 8 + (hi << 2));
      o1 = wmma_bf16(pa, vf, o1);
      vf.q[0] = *(const uint4*)(vr2 + (hi << 2));
      vf.q[1] = *(const uint4*)(vr2 + 8 + (hi << 2));
      o2 = wmma_bf16(pa, vf, o2);
      vf.q[0] = *(const uint4*)(vr3 + (hi << 2));
      vf.q[1] = *(const uint4*)(vr3 + 8 + (hi << 2));
      o3 = wmma_bf16(pa, vf, o3);
    }
  }

#pragma unroll
  for (int r = 0; r < 8; ++r) {
    const int l = l0 + r + (hi << 3);
    if (l < Sq) {
      const float inv = 1.0f / lrun[r];
      const size_t base = ((size_t)(b * Sq + l)) * Ee + (size_t)h * DhD;
      O[base + 0  + m] = f2bf(o0[r] * inv);
      O[base + 16 + m] = f2bf(o1[r] * inv);
      O[base + 32 + m] = f2bf(o2[r] * inv);
      O[base + 48 + m] = f2bf(o3[r] * inv);
    }
  }
}

// ---------------- LayerNorm over E=512, in-place fp32 + bf16 mirror ----------------
__global__ __launch_bounds__(256) void ln_kernel(float* xf, unsigned short* __restrict__ xh,
                                                 const float* __restrict__ w,
                                                 const float* __restrict__ bb)
{
  const int row = blockIdx.x;
  __shared__ float red[256];
  float* xr = xf + (size_t)row * Ee;
  float s = 0.0f;
  for (int e = threadIdx.x; e < Ee; e += 256) s += xr[e];
  red[threadIdx.x] = s; __syncthreads();
  for (int off = 128; off > 0; off >>= 1) {
    if (threadIdx.x < off) red[threadIdx.x] += red[threadIdx.x + off];
    __syncthreads();
  }
  const float mu = red[0] * (1.0f / Ee);
  __syncthreads();
  float v = 0.0f;
  for (int e = threadIdx.x; e < Ee; e += 256) { const float d = xr[e] - mu; v += d * d; }
  red[threadIdx.x] = v; __syncthreads();
  for (int off = 128; off > 0; off >>= 1) {
    if (threadIdx.x < off) red[threadIdx.x] += red[threadIdx.x + off];
    __syncthreads();
  }
  const float rstd = rsqrtf(red[0] * (1.0f / Ee) + 1e-5f);
  for (int e = threadIdx.x; e < Ee; e += 256) {
    const float o = (xr[e] - mu) * rstd * w[e] + bb[e];
    xr[e] = o;
    xh[(size_t)row * Ee + e] = f2bf(o);
  }
}

// ---------------- discriminator: wave-per-token dot over 2F=1536 ----------------
__global__ __launch_bounds__(256) void disc_kernel(const float* __restrict__ bert,
                                                   const float* __restrict__ pf,
                                                   const float* __restrict__ wd,
                                                   const float* __restrict__ bd,
                                                   float* __restrict__ outp)
{
  const int wave = threadIdx.x >> 5, lane = threadIdx.x & 31;
  const int t = blockIdx.x * 8 + wave;
  if (t >= Tt) return;
  const float* br = bert + (size_t)t * Ff;
  const float* pr = pf + (size_t)t * Ff;
  float s = 0.0f;
  for (int f = lane; f < Ff; f += 32) s += br[f] * wd[f] + pr[f] * wd[Ff + f];
  for (int off = 16; off; off >>= 1) s += __shfl_xor(s, off, 32);
  if (lane == 0) outp[t] = 1.0f / (1.0f + __expf(-(s + bd[0])));
}

// ---------------- host orchestration ----------------
extern "C" void kernel_launch(void* const* d_in, const int* in_sizes, int n_in,
                              void* d_out, int out_size, void* d_ws, size_t ws_size,
                              hipStream_t stream) {
  (void)in_sizes; (void)n_in; (void)out_size; (void)ws_size;
  const int*   pos1  = (const int*)d_in[0];
  const int*   pos2  = (const int*)d_in[1];
  const float* bert  = (const float*)d_in[2];
  const float* fbase = (const float*)d_in[3];
  const float* emb   = (const float*)d_in[4];
  const float* inw   = (const float*)d_in[5];
  const float* inb   = (const float*)d_in[6];
  const float* outw  = (const float*)d_in[7];
  const float* outb  = (const float*)d_in[8];
  const float* ln1w  = (const float*)d_in[9];
  const float* ln1b  = (const float*)d_in[10];
  const float* ln2w  = (const float*)d_in[11];
  const float* ln2b  = (const float*)d_in[12];
  const float* w1    = (const float*)d_in[13];
  const float* b1    = (const float*)d_in[14];
  const float* w2    = (const float*)d_in[15];
  const float* b2    = (const float*)d_in[16];
  const float* wm    = (const float*)d_in[17];
  const float* wd    = (const float*)d_in[18];
  const float* bd    = (const float*)d_in[19];

  char* wp = (char*)d_ws;
  auto grab = [&](size_t bytes) -> void* {
    void* r = (void*)wp; wp += (bytes + 255) & ~(size_t)255; return r;
  };
  float*          xf  = (float*)grab((size_t)Tt * Ee * 4);
  unsigned short* xh  = (unsigned short*)grab((size_t)Tt * Ee * 2);
  unsigned short* xo  = (unsigned short*)grab((size_t)Tt * Ee * 2);
  unsigned short* hb  = (unsigned short*)grab((size_t)Tt * DFFd * 2);
  float*          pf  = (float*)grab((size_t)Tt * Ff * 4);
  unsigned short* Qb  = (unsigned short*)grab((size_t)BH * Sq * DhD * 2);
  unsigned short* Kb  = (unsigned short*)grab((size_t)BH * SPp * DhD * 2);
  unsigned short* Vtb = (unsigned short*)grab((size_t)BH * DhD * SPp * 2);
  unsigned short* wInB  = (unsigned short*)grab((size_t)2 * 3 * Ee * Ee * 2);
  unsigned short* wOutB = (unsigned short*)grab((size_t)2 * Ee * Ee * 2);
  unsigned short* wF1B  = (unsigned short*)grab((size_t)2 * DFFd * Ee * 2);
  unsigned short* wF2B  = (unsigned short*)grab((size_t)2 * Ee * DFFd * 2);
  unsigned short* wMB   = (unsigned short*)grab((size_t)Ff * Ee * 2);

  auto conv = [&](const float* src, unsigned short* dst, size_t n) {
    f2bf_kernel<<<dim3((unsigned)((n + 255) / 256)), dim3(256), 0, stream>>>(src, dst, (int)n);
  };
  conv(inw,  wInB,  (size_t)2 * 3 * Ee * Ee);
  conv(outw, wOutB, (size_t)2 * Ee * Ee);
  conv(w1,   wF1B,  (size_t)2 * DFFd * Ee);
  conv(w2,   wF2B,  (size_t)2 * Ee * DFFd);
  conv(wm,   wMB,   (size_t)Ff * Ee);

  embed_kernel<<<dim3(Tt), dim3(128), 0, stream>>>(pos1, pos2, fbase, emb, xf, xh);
  (void)hipMemsetAsync(Kb,  0, (size_t)BH * SPp * DhD * 2, stream);   // zero pad rows
  (void)hipMemsetAsync(Vtb, 0, (size_t)BH * DhD * SPp * 2, stream);   // zero pad cols

  const dim3 blk(256);
  const size_t smem512  = (size_t)16 * Ee * 2;     // 16 KB A strip (K=512)
  const size_t smem2048 = (size_t)16 * DFFd * 2;   // 64 KB A strip (K=2048)
  for (int l = 0; l < 2; ++l) {
    // QKV projection, routed into Q / padded-K / transposed-V (bf16)
    gemm_kernel<<<dim3(6, Tt / 16), blk, smem512, stream>>>(
        xh, wInB + (size_t)l * 3 * Ee * Ee, inb + (size_t)l * 3 * Ee,
        nullptr, nullptr, nullptr, Qb, Kb, Vtb, Ee, 3 * Ee, 3);
    // flash attention -> xo (bf16 [T,E])
    attn_kernel<<<dim3(33, BH), dim3(32), 0, stream>>>(Qb, Kb, Vtb, xo);
    // out_proj + residual into fp32 master
    gemm_kernel<<<dim3(2, Tt / 16), blk, smem512, stream>>>(
        xo, wOutB + (size_t)l * Ee * Ee, outb + (size_t)l * Ee,
        xf, xf, nullptr, nullptr, nullptr, nullptr, Ee, Ee, 0);
    ln_kernel<<<dim3(Tt), blk, 0, stream>>>(xf, xh, ln1w + (size_t)l * Ee, ln1b + (size_t)l * Ee);
    // FFN1 (ReLU -> bf16 hidden)
    gemm_kernel<<<dim3(8, Tt / 16), blk, smem512, stream>>>(
        xh, wF1B + (size_t)l * DFFd * Ee, b1 + (size_t)l * DFFd,
        nullptr, nullptr, hb, nullptr, nullptr, nullptr, Ee, DFFd, 1);
    // FFN2 + residual
    gemm_kernel<<<dim3(2, Tt / 16), blk, smem2048, stream>>>(
        hb, wF2B + (size_t)l * Ee * DFFd, b2 + (size_t)l * Ee,
        xf, xf, nullptr, nullptr, nullptr, nullptr, DFFd, Ee, 0);
    ln_kernel<<<dim3(Tt), blk, 0, stream>>>(xf, xh, ln2w + (size_t)l * Ee, ln2b + (size_t)l * Ee);
  }

  // pos_feature = tanh(x @ w_model^T)  (fp32)
  gemm_kernel<<<dim3(3, Tt / 16), blk, smem512, stream>>>(
      xh, wMB, nullptr, nullptr, pf, nullptr, nullptr, nullptr, nullptr, Ee, Ff, 2);
  // sigmoid(concat(bert, pos) @ w_disc^T + b)
  disc_kernel<<<dim3(Tt / 8), blk, 0, stream>>>(bert, pf, wd, bd, (float*)d_out);
}
